// Head_33861522162444
// MI455X (gfx1250) — matrix operand order repaired and checked by reading, not compiled, not run on hardware
//
#include <hip/hip_runtime.h>
#include <hip/hip_bf16.h>

typedef __bf16 bf16_t;
typedef __attribute__((ext_vector_type(16))) __bf16 v16bf;
typedef __attribute__((ext_vector_type(8)))  __bf16 v8bf;
typedef __attribute__((ext_vector_type(8)))  float  v8f;

#define HEAD  128
#define NEMBD 1024
#define TSEQ  2048
#define NBAT  8

static __device__ __forceinline__ bf16_t f2bf(float f) {
  // round-to-nearest-even f32 -> bf16 via bit trick
  unsigned u = __float_as_uint(f);
  u = u + 0x7FFFu + ((u >> 16) & 1u);
  unsigned short h = (unsigned short)(u >> 16);
  bf16_t b;
  __builtin_memcpy(&b, &h, sizeof(h));
  return b;
}

static __device__ __forceinline__ v8f wmma_bf16(v16bf a, v16bf b, v8f c) {
  return __builtin_amdgcn_wmma_f32_16x16x32_bf16(false, a, false, b,
                                                 (short)0, c, false, false);
}

// Async 16-byte global -> LDS copy (per-lane), tracked by ASYNCcnt.
static __device__ __forceinline__ void async_ld16(unsigned lds_byte_off,
                                                  const void* gptr) {
  asm volatile("global_load_async_to_lds_b128 %0, %1, off"
               :: "v"(lds_byte_off), "v"((unsigned long long)gptr)
               : "memory");
}
static __device__ __forceinline__ void wait_async0() {
  asm volatile("s_wait_asynccnt 0x0" ::: "memory");
}

// ---------------------------------------------------------------------------
// Kernel 0: convert W to bf16, transposed [mat][H][D]; fold scale 1/32 into Wq
// ---------------------------------------------------------------------------
__global__ void wmmaattn_convert_w(const float* __restrict__ Wq,
                                   const float* __restrict__ Wk,
                                   const float* __restrict__ Wv,
                                   bf16_t* __restrict__ WbT) {
  const int per = NEMBD * HEAD;  // 131072
  int i = blockIdx.x * blockDim.x + threadIdx.x;
  if (i >= 3 * per) return;
  int mat = i / per;
  int off = i - mat * per;       // off = d*HEAD + h
  int d = off >> 7;
  int h = off & 127;
  float v;
  if (mat == 0)      v = Wq[off] * 0.03125f;   // fold n_embd^-0.5
  else if (mat == 1) v = Wk[off];
  else               v = Wv[off];
  WbT[(size_t)mat * per + (size_t)h * NEMBD + d] = f2bf(v);
}

// ---------------------------------------------------------------------------
// Kernel 1: QKV projection.  Block = 128 thr (4 waves), 16 rows, 24 N-tiles.
// x tiles double-buffered in LDS via async global->LDS copies.
// Dynamic smem layout: two 16x32 f32 tiles at bytes [0,2048) and [2048,4096).
// ---------------------------------------------------------------------------
__global__ void __launch_bounds__(128)
wmmaattn_qkv(const float* __restrict__ x, const bf16_t* __restrict__ WbT,
             bf16_t* __restrict__ qb, bf16_t* __restrict__ kb,
             bf16_t* __restrict__ vb) {
  extern __shared__ char smem[];
  const int tid  = threadIdx.x;
  const int wave = tid >> 5;
  const int lane = tid & 31;
  const int half = lane >> 4;
  const int nl   = lane & 15;
  const size_t row0 = (size_t)blockIdx.x * 16;

  v8f acc[6];
#pragma unroll
  for (int t = 0; t < 6; ++t)
#pragma unroll
    for (int r = 0; r < 8; ++r) acc[t][r] = 0.0f;

  const int xr = tid >> 3;          // row within 16x32 tile
  const int xc = (tid & 7) * 4;     // col within 16x32 tile (16B chunk)

  // prefetch k-step 0 into buffer 0
  async_ld16((unsigned)(tid * 16), x + (row0 + xr) * NEMBD + xc);

  for (int ks = 0; ks < 32; ++ks) {
    wait_async0();
    __syncthreads();
    if (ks + 1 < 32)   // prefetch next k-step into the other buffer
      async_ld16((unsigned)(((ks + 1) & 1) * 2048 + tid * 16),
                 x + (row0 + xr) * NEMBD + (ks + 1) * 32 + xc);

    const float* X = (const float*)(smem + (ks & 1) * 2048);
    // A fragment (16x32 bf16): lanes 0-15 K0-15, lanes 16-31 K16-31
    v16bf a;
#pragma unroll
    for (int e = 0; e < 16; ++e) {
      int kk = 16 * half + e;
      a[e] = f2bf(X[nl * 32 + kk]);
    }
    const int k0 = ks * 32;
#pragma unroll
    for (int t = 0; t < 6; ++t) {
      int nt  = wave * 6 + t;          // 0..23 over {q,k,v} x 8 col-tiles
      int mat = nt >> 3;
      int col = (nt & 7) * 16;
      const bf16_t* wp = WbT + (size_t)mat * (NEMBD * HEAD)
                       + (size_t)(col + nl) * NEMBD + k0 + 16 * half;
      v16bf b = *(const v16bf*)wp;     // contiguous from transposed W
      acc[t] = wmma_bf16(a, b, acc[t]);
    }
  }

#pragma unroll
  for (int t = 0; t < 6; ++t) {
    int nt  = wave * 6 + t;
    int mat = nt >> 3;
    int col = (nt & 7) * 16;
    bf16_t* op = (mat == 0) ? qb : (mat == 1) ? kb : vb;
#pragma unroll
    for (int r = 0; r < 8; ++r)
      op[(row0 + r + 8 * half) * HEAD + col + nl] = f2bf(acc[t][r]);
  }
}

// ---------------------------------------------------------------------------
// Kernel 2: flash attention.  Block = 64 query rows (4 waves x 16-row tile),
// key blocks of 32, online softmax, double-buffered async K/V staging.
// Dynamic smem layout (bytes):
//   K buffers : [0,8192) and [8192,16384)
//   V buffers : [16384,24576) and [24576,32768)   (row-major)
//   P scratch : [32768 + wave*1024, +1024)         (16x32 bf16 per wave)
// ---------------------------------------------------------------------------
static __device__ __forceinline__ void stage_kv_async(
    int tid, int buf, int key0, const bf16_t* kB, const bf16_t* vB) {
  const int kr = tid >> 2;             // key row 0..31
  const int kc = (tid & 3) * 32;       // 64B span -> 4 x 16B chunks
  const unsigned lk = (unsigned)(buf * 8192 + (kr * HEAD + kc) * 2);
  const bf16_t* gk = kB + (size_t)(key0 + kr) * HEAD + kc;
  const unsigned lv = (unsigned)(16384 + buf * 8192 + (kr * HEAD + kc) * 2);
  const bf16_t* gv = vB + (size_t)(key0 + kr) * HEAD + kc;
#pragma unroll
  for (int i = 0; i < 4; ++i) {
    async_ld16(lk + 16 * i, gk + 8 * i);
    async_ld16(lv + 16 * i, gv + 8 * i);
  }
}

__global__ void __launch_bounds__(128)
wmmaattn_fa(const bf16_t* __restrict__ qb, const bf16_t* __restrict__ kb,
            const bf16_t* __restrict__ vb, float* __restrict__ out) {
  extern __shared__ char smem[];

  const int tid   = threadIdx.x;
  const int wave  = tid >> 5;
  const int lane  = tid & 31;
  const int half  = lane >> 4;
  const int nl    = lane & 15;
  const int batch = blockIdx.y;
  const int rb    = blockIdx.x;                  // 32 row-blocks of 64
  const int qrow0 = rb * 64 + wave * 16;

  const size_t base = (size_t)batch * TSEQ * HEAD;
  const bf16_t* qB = qb + base;
  const bf16_t* kB = kb + base;
  const bf16_t* vB = vb + base;
  float* oB = out + base;

  // preload Q row tile as 4 A-fragments (K-steps of 32 over HEAD=128)
  v16bf qa[4];
#pragma unroll
  for (int c = 0; c < 4; ++c) {
    const bf16_t* qp = qB + (size_t)(qrow0 + nl) * HEAD + 32 * c;
    v8bf lo = *(const v8bf*)(qp + 16 * half);
    v8bf hi = *(const v8bf*)(qp + 16 * half + 8);
    qa[c] = __builtin_shufflevector(lo, hi, 0,1,2,3,4,5,6,7,
                                            8,9,10,11,12,13,14,15);
  }

  v8f acc[8];
  float mrow[8], lrow[8];
#pragma unroll
  for (int r = 0; r < 8; ++r) { mrow[r] = -__builtin_inff(); lrow[r] = 0.0f; }
#pragma unroll
  for (int nt = 0; nt < 8; ++nt)
#pragma unroll
    for (int r = 0; r < 8; ++r) acc[nt][r] = 0.0f;

  const int nkb = (rb + 1) * 2;            // causal: key blocks 0..rb*64+63
  stage_kv_async(tid, 0, 0, kB, vB);       // prefetch block 0

  for (int j = 0; j < nkb; ++j) {
    const int key0 = j * 32;
    wait_async0();
    __syncthreads();
    if (j + 1 < nkb)                       // prefetch next block, other buffer
      stage_kv_async(tid, (j + 1) & 1, (j + 1) * 32, kB, vB);

    if (key0 <= qrow0 + 15) {              // wave-uniform: EXEC stays full
      const bf16_t* Kl = (const bf16_t*)(smem + (j & 1) * 8192);
      const bf16_t* Vl = (const bf16_t*)(smem + 16384 + (j & 1) * 8192);
      bf16_t*       P  = (bf16_t*)(smem + 32768 + wave * 1024);

      v8f s0, s1;
#pragma unroll
      for (int r = 0; r < 8; ++r) { s0[r] = 0.0f; s1[r] = 0.0f; }
#pragma unroll
      for (int c = 0; c < 4; ++c) {
        v16bf b0 = *(const v16bf*)&Kl[nl * HEAD + 32 * c + 16 * half];
        v16bf b1 = *(const v16bf*)&Kl[(16 + nl) * HEAD + 32 * c + 16 * half];
        s0 = wmma_bf16(qa[c], b0, s0);
        s1 = wmma_bf16(qa[c], b1, s1);
      }
      if (key0 + 31 > qrow0) {             // causal mask on straddling blocks
#pragma unroll
        for (int r = 0; r < 8; ++r) {
          int mg = qrow0 + r + 8 * half;
          s0[r] = (key0 + nl      > mg) ? -__builtin_inff() : s0[r];
          s1[r] = (key0 + 16 + nl > mg) ? -__builtin_inff() : s1[r];
        }
      }
      float alph[8];
#pragma unroll
      for (int r = 0; r < 8; ++r) {
        float mx = fmaxf(s0[r], s1[r]);
        mx = fmaxf(mx, __shfl_xor(mx, 1, 32));
        mx = fmaxf(mx, __shfl_xor(mx, 2, 32));
        mx = fmaxf(mx, __shfl_xor(mx, 4, 32));
        mx = fmaxf(mx, __shfl_xor(mx, 8, 32));
        float mn = fmaxf(mrow[r], mx);
        float a  = __expf(mrow[r] - mn);
        mrow[r]  = mn;
        float p0 = __expf(s0[r] - mn);
        float p1 = __expf(s1[r] - mn);
        s0[r] = p0; s1[r] = p1;
        float rs = p0 + p1;
        rs += __shfl_xor(rs, 1, 32);
        rs += __shfl_xor(rs, 2, 32);
        rs += __shfl_xor(rs, 4, 32);
        rs += __shfl_xor(rs, 8, 32);
        lrow[r] = lrow[r] * a + rs;
        alph[r] = a;
      }
#pragma unroll
      for (int nt = 0; nt < 8; ++nt)
#pragma unroll
        for (int r = 0; r < 8; ++r) acc[nt][r] *= alph[r];

      // C-layout -> A-layout for P via per-wave LDS scratch
#pragma unroll
      for (int r = 0; r < 8; ++r) {
        P[(r + 8 * half) * 32 + nl]      = f2bf(s0[r]);
        P[(r + 8 * half) * 32 + 16 + nl] = f2bf(s1[r]);
      }
      __asm__ volatile("s_wait_dscnt 0" ::: "memory");
      v8bf plo = *(const v8bf*)&P[nl * 32 + 8 * half];
      v8bf phi = *(const v8bf*)&P[nl * 32 + 16 + 8 * half];
      v16bf pa = __builtin_shufflevector(plo, phi, 0,1,2,3,4,5,6,7,
                                                   8,9,10,11,12,13,14,15);
#pragma unroll
      for (int nt = 0; nt < 8; ++nt) {
        // B fragment of V (row-major in LDS): strided gather per lane
        v16bf bv;
#pragma unroll
        for (int e = 0; e < 16; ++e)
          bv[e] = Vl[((e < 8) ? (8 * half + e) : (16 + 8 * half + (e - 8))) * HEAD + nt * 16 + nl];
        acc[nt] = wmma_bf16(pa, bv, acc[nt]);
      }
    }
  }

#pragma unroll
  for (int r = 0; r < 8; ++r) {
    float inv = 1.0f / lrow[r];
    size_t rg = (size_t)(qrow0 + r + 8 * half);
#pragma unroll
    for (int nt = 0; nt < 8; ++nt)
      oB[rg * HEAD + nt * 16 + nl] = acc[nt][r] * inv;
  }
}

// ---------------------------------------------------------------------------
extern "C" void kernel_launch(void* const* d_in, const int* in_sizes, int n_in,
                              void* d_out, int out_size, void* d_ws, size_t ws_size,
                              hipStream_t stream) {
  const float* x  = (const float*)d_in[0];
  const float* Wq = (const float*)d_in[1];
  const float* Wk = (const float*)d_in[2];
  const float* Wv = (const float*)d_in[3];
  float* out = (float*)d_out;

  char* ws = (char*)d_ws;
  bf16_t* WbT = (bf16_t*)ws;                                // 768 KB
  bf16_t* qb  = (bf16_t*)(ws + 786432);                     // 4 MB each
  bf16_t* kb  = (bf16_t*)(ws + 786432 + 4194304);
  bf16_t* vb  = (bf16_t*)(ws + 786432 + 2 * 4194304);

  wmmaattn_convert_w<<<dim3((3 * NEMBD * HEAD + 255) / 256), dim3(256), 0,
                       stream>>>(Wq, Wk, Wv, WbT);
  wmmaattn_qkv<<<dim3((NBAT * TSEQ) / 16), dim3(128), 4096, stream>>>(
      x, WbT, qb, kb, vb);
  wmmaattn_fa<<<dim3(TSEQ / 64, NBAT), dim3(128), 36864, stream>>>(qb, kb, vb,
                                                                   out);
}
